// Encoder_75668733821146
// MI455X (gfx1250) — matrix operand order repaired and checked by reading, not compiled, not run on hardware
//
#include <hip/hip_runtime.h>
#include <math.h>

#define BN_EPS 1e-5f

typedef __bf16 v16bf __attribute__((ext_vector_type(16)));
typedef float  v8f   __attribute__((ext_vector_type(8)));

union BF16Frag { v16bf v; __bf16 b[16]; uint4 q[2]; };

__device__ __forceinline__ float gelu_exact(float x) {
    return 0.5f * x * (1.0f + erff(x * 0.70710678118654752f));
}
__device__ __forceinline__ float sigmoidf_(float x) {
    return 1.0f / (1.0f + expf(-x));
}
__host__ __device__ __forceinline__ int imin(int a, int b) { return a < b ? a : b; }

__device__ __forceinline__ float4 ld4(const float* p) { return *(const float4*)p; }
__device__ __forceinline__ float4 bn4(float4 v, float4 s, float4 t) {
    return make_float4(v.x * s.x + t.x, v.y * s.y + t.y, v.z * s.z + t.z, v.w * s.w + t.w);
}
__device__ __forceinline__ float4 mul4(float4 v, float s) {
    return make_float4(v.x * s, v.y * s, v.z * s, v.w * s);
}
__device__ __forceinline__ void cvt4(__bf16* d, float4 v) {
    d[0] = (__bf16)v.x; d[1] = (__bf16)v.y; d[2] = (__bf16)v.z; d[3] = (__bf16)v.w;
}

// ---------------- utility fills ----------------
__global__ void k_fill_f32(float* p, long long n, float v) {
    long long i = (long long)blockIdx.x * blockDim.x + threadIdx.x;
    long long st = (long long)gridDim.x * blockDim.x;
    for (; i < n; i += st) p[i] = v;
}
__global__ void k_scale_f32(float* p, long long n, float s) {
    long long i = (long long)blockIdx.x * blockDim.x + threadIdx.x;
    long long st = (long long)gridDim.x * blockDim.x;
    for (; i < n; i += st) p[i] *= s;
}

// ---------------- BatchNorm stats ----------------
__global__ void k_bn_stats(const float* __restrict__ h, float* __restrict__ sums,
                           long long rows) {
    __shared__ float s1[64], s2[64];
    if (threadIdx.x < 64) { s1[threadIdx.x] = 0.f; s2[threadIdx.x] = 0.f; }
    __syncthreads();
    int c = threadIdx.x & 63;
    long long r = (long long)blockIdx.x * (blockDim.x >> 6) + (threadIdx.x >> 6);
    long long st = (long long)gridDim.x * (blockDim.x >> 6);
    float a = 0.f, b = 0.f;
    for (; r < rows; r += st) {
        float v = h[r * 64 + c];
        a += v; b += v * v;
    }
    atomicAdd(&s1[c], a); atomicAdd(&s2[c], b);
    __syncthreads();
    if (threadIdx.x < 64) {
        atomicAdd(&sums[threadIdx.x], s1[threadIdx.x]);
        atomicAdd(&sums[64 + threadIdx.x], s2[threadIdx.x]);
    }
}

__global__ void k_bn_finalize(const float* sums, const float* gamma, const float* beta,
                              float* scale, float* shift, float invNT) {
    int j = threadIdx.x;
    if (j >= 64) return;
    float mu  = sums[j] * invNT;
    float var = sums[64 + j] * invNT - mu * mu;
    float sc  = gamma[j] * rsqrtf(var + BN_EPS);
    scale[j] = sc;
    shift[j] = beta[j] - mu * sc;
}

// ---------------- degrees / norms / CSR ----------------
__global__ void k_degrees(const int* __restrict__ src, const int* __restrict__ dst, int E,
                          int* outdeg, int* indeg) {
    int i = blockIdx.x * blockDim.x + threadIdx.x;
    int st = gridDim.x * blockDim.x;
    for (; i < E; i += st) {
        atomicAdd(&outdeg[src[i]], 1);
        atomicAdd(&indeg[dst[i]], 1);
    }
}
__global__ void k_norms(const int* outdeg, const int* indeg, float* onorm, float* inorm, int N) {
    int i = blockIdx.x * blockDim.x + threadIdx.x;
    if (i >= N) return;
    onorm[i] = rsqrtf(fmaxf((float)outdeg[i], 1.0f));
    inorm[i] = rsqrtf(fmaxf((float)indeg[i], 1.0f));
}
__global__ void k_scan(const int* __restrict__ cnt, int* __restrict__ off, int N) {
    __shared__ int part[256];
    __shared__ int totalS;
    int tid = threadIdx.x;
    int chunk = (N + 255) / 256;
    int beg = tid * chunk, end = imin(beg + chunk, N);
    int s = 0;
    for (int i = beg; i < end; ++i) s += cnt[i];
    part[tid] = s;
    __syncthreads();
    if (tid == 0) {
        int run = 0;
        for (int i = 0; i < 256; ++i) { int v = part[i]; part[i] = run; run += v; }
        totalS = run;
    }
    __syncthreads();
    int run = part[tid];
    for (int i = beg; i < end; ++i) { off[i] = run; run += cnt[i]; }
    if (tid == 0) off[N] = totalS;
}
__global__ void k_csr_fill(const int* __restrict__ src, const int* __restrict__ dst, int E,
                           const int* __restrict__ off, int* __restrict__ cur,
                           int* __restrict__ csrsrc) {
    int i = blockIdx.x * blockDim.x + threadIdx.x;
    int st = gridDim.x * blockDim.x;
    for (; i < E; i += st) {
        int d = dst[i];
        int pos = atomicAdd(&cur[d], 1);
        csrsrc[off[d] + pos] = src[i];
    }
}

// ---------------- weight packing into WMMA B-operand layout ----------------
// Packed index: ((tile*KS + ks)*32 + lane)*16 + e ; element holds W[K][col] with
// K = 32*ks + 8*(lane>>4) + (e&7) + 16*(e>>3) , col = 16*tile + (lane&15)
__global__ void k_pack_w64(const float* __restrict__ W, __bf16* __restrict__ Wp) {
    int idx = blockIdx.x * blockDim.x + threadIdx.x;  // 4096 total
    if (idx >= 4096) return;
    int e = idx & 15, lane = (idx >> 4) & 31, ks = (idx >> 9) & 1, nt = idx >> 10;
    int kh = lane >> 4;
    int K = 32 * ks + 8 * kh + (e & 7) + 16 * (e >> 3);
    int col = 16 * nt + (lane & 15);
    Wp[idx] = (__bf16)W[K * 64 + col];
}
__global__ void k_pack_lstm(const float* __restrict__ Wih, const float* __restrict__ Whh,
                            __bf16* __restrict__ Wp) {
    int idx = blockIdx.x * blockDim.x + threadIdx.x;  // 32768 total
    if (idx >= 32768) return;
    int e = idx & 15, lane = (idx >> 4) & 31, ks = (idx >> 9) & 3, ct = idx >> 11;
    int kh = lane >> 4;
    int K = 32 * ks + 8 * kh + (e & 7) + 16 * (e >> 3);  // 0..127
    int col = 16 * ct + (lane & 15);         // 0..255 (gate output index)
    float v = (K < 64) ? Wih[col * 64 + K] : Whh[col * 64 + (K - 64)];
    Wp[idx] = (__bf16)v;
}

// ---------------- spatial GEMM: M = (A*onorm) @ W  (A: N x 64, W: 64 x 64) ------------
// mode 0: A = h slice (row stride = T*64), apply BN scale/shift first.
// mode 1: A = plain N x 64 buffer (row stride 64).
__global__ void k_gemm64(const float* __restrict__ A, const float* __restrict__ scale,
                         const float* __restrict__ shift, const float* __restrict__ onorm,
                         const __bf16* __restrict__ Wp, float* __restrict__ out,
                         int N, int rowStride, int mode) {
    int wave = blockIdx.x * (blockDim.x >> 5) + (threadIdx.x >> 5);
    int rowBase = wave * 16;
    if (rowBase >= N) return;
    int lane = threadIdx.x & 31;
    int m = lane & 15, kh = lane >> 4;
    int rc = imin(rowBase + m, N - 1);
    const float* arow = A + (size_t)rc * rowStride;
    float on = onorm[rc];
    BF16Frag a[2];
    #pragma unroll
    for (int ks = 0; ks < 2; ++ks) {
        int base = 32 * ks + 8 * kh;
        float4 x0 = ld4(arow + base),      x1 = ld4(arow + base + 4);
        float4 y0 = ld4(arow + base + 16), y1 = ld4(arow + base + 20);
        if (mode == 0) {
            x0 = bn4(x0, ld4(scale + base),      ld4(shift + base));
            x1 = bn4(x1, ld4(scale + base + 4),  ld4(shift + base + 4));
            y0 = bn4(y0, ld4(scale + base + 16), ld4(shift + base + 16));
            y1 = bn4(y1, ld4(scale + base + 20), ld4(shift + base + 20));
        }
        cvt4(&a[ks].b[0],  mul4(x0, on));
        cvt4(&a[ks].b[4],  mul4(x1, on));
        cvt4(&a[ks].b[8],  mul4(y0, on));
        cvt4(&a[ks].b[12], mul4(y1, on));
    }
    bool fullTile = (rowBase + 16 <= N);    // wave-uniform
    for (int nt = 0; nt < 4; ++nt) {
        v8f c = {};
        #pragma unroll
        for (int ks = 0; ks < 2; ++ks) {
            const uint4* bp = (const uint4*)(Wp + ((size_t)(nt * 2 + ks) * 32 + lane) * 16);
            BF16Frag bf; bf.q[0] = bp[0]; bf.q[1] = bp[1];
            c = __builtin_amdgcn_wmma_f32_16x16x32_bf16(false, a[ks].v, false, bf.v,
                                                        (short)0, c, false, false);
        }
        int col = 16 * nt + (lane & 15);
        float* orow = out + (size_t)(rowBase + 8 * kh) * 64 + col;
        if (fullTile) {
            #pragma unroll
            for (int r = 0; r < 8; ++r) orow[r * 64] = c[r];
        } else {
            #pragma unroll
            for (int r = 0; r < 8; ++r)
                if (rowBase + 8 * kh + r < N) orow[r * 64] = c[r];
        }
    }
}

// ---------------- CSR gather + GELU (conv epilogue), one wave per node --------------
__global__ void k_agg_gelu(const float* __restrict__ Min, const int* __restrict__ off,
                           const int* __restrict__ csrsrc, const float* __restrict__ inorm,
                           const float* __restrict__ bias, float* __restrict__ Yout, int N) {
    int node = blockIdx.x * (blockDim.x >> 5) + (threadIdx.x >> 5);
    if (node >= N) return;
    int lane = threadIdx.x & 31;
    int b = off[node], e = off[node + 1];
    float a0 = 0.f, a1 = 0.f;
    if (b < e) {
        int idx = csrsrc[b];
        for (int k = b; k < e; ++k) {
            const float* mr = Min + (size_t)idx * 64;
            int nidx = (k + 1 < e) ? csrsrc[k + 1] : idx;
            __builtin_prefetch(Min + (size_t)nidx * 64 + 2 * lane, 0, 0);
            a0 += mr[lane]; a1 += mr[lane + 32];
            idx = nidx;
        }
    }
    float inn = inorm[node];
    Yout[(size_t)node * 64 + lane]      = gelu_exact(a0 * inn + bias[lane]);
    Yout[(size_t)node * 64 + lane + 32] = gelu_exact(a1 * inn + bias[lane + 32]);
}

// second conv layer fused with node-mean readout (accumulated into hs via atomics)
__global__ void k_agg_gelu_readout(const float* __restrict__ Min, const int* __restrict__ off,
                                   const int* __restrict__ csrsrc, const float* __restrict__ inorm,
                                   const float* __restrict__ bias, float* __restrict__ hs_out,
                                   int N) {
    __shared__ float part[64];
    if (threadIdx.x < 64) part[threadIdx.x] = 0.f;
    __syncthreads();
    int node = blockIdx.x * (blockDim.x >> 5) + (threadIdx.x >> 5);
    int lane = threadIdx.x & 31;
    float g0 = 0.f, g1 = 0.f;
    if (node < N) {
        int b = off[node], e = off[node + 1];
        float a0 = 0.f, a1 = 0.f;
        if (b < e) {
            int idx = csrsrc[b];
            for (int k = b; k < e; ++k) {
                const float* mr = Min + (size_t)idx * 64;
                int nidx = (k + 1 < e) ? csrsrc[k + 1] : idx;
                __builtin_prefetch(Min + (size_t)nidx * 64 + 2 * lane, 0, 0);
                a0 += mr[lane]; a1 += mr[lane + 32];
                idx = nidx;
            }
        }
        float inn = inorm[node];
        g0 = gelu_exact(a0 * inn + bias[lane]);
        g1 = gelu_exact(a1 * inn + bias[lane + 32]);
    }
    atomicAdd(&part[lane], g0);
    atomicAdd(&part[lane + 32], g1);
    __syncthreads();
    if (threadIdx.x < 64) atomicAdd(&hs_out[threadIdx.x], part[threadIdx.x]);
}

// ---------------- LSTM gates GEMM: gates = [xt | h_prev] @ [Wih^T ; Whh^T] ----------
// wave handles 16 rows x 64 cols (one gate quarter); K = 128 (4 WMMA k-steps)
__global__ void k_lstm_gemm(const float* __restrict__ h, int t, int T,
                            const float* __restrict__ scale, const float* __restrict__ shift,
                            const float* __restrict__ hstate,
                            const __bf16* __restrict__ Wp,
                            float* __restrict__ gates, int N) {
    int wib = threadIdx.x >> 5;
    int lane = threadIdx.x & 31;
    int mt = wib & 1;
    int cq = wib >> 1;                 // gate quarter 0..3
    int rowBase = blockIdx.x * 32 + mt * 16;
    if (rowBase >= N) return;
    int m = lane & 15, kh = lane >> 4;
    int rc = imin(rowBase + m, N - 1);
    const float* xr = h + (size_t)rc * (T * 64) + (size_t)t * 64;
    const float* hr = hstate + (size_t)rc * 64;
    BF16Frag a[4];
    #pragma unroll
    for (int ks = 0; ks < 4; ++ks) {
        const float* sp = (ks < 2) ? xr : hr;
        int base = 32 * (ks & 1) + 8 * kh;
        float4 x0 = ld4(sp + base),      x1 = ld4(sp + base + 4);
        float4 y0 = ld4(sp + base + 16), y1 = ld4(sp + base + 20);
        if (ks < 2) {
            x0 = bn4(x0, ld4(scale + base),      ld4(shift + base));
            x1 = bn4(x1, ld4(scale + base + 4),  ld4(shift + base + 4));
            y0 = bn4(y0, ld4(scale + base + 16), ld4(shift + base + 16));
            y1 = bn4(y1, ld4(scale + base + 20), ld4(shift + base + 20));
        }
        cvt4(&a[ks].b[0], x0);  cvt4(&a[ks].b[4], x1);
        cvt4(&a[ks].b[8], y0);  cvt4(&a[ks].b[12], y1);
    }
    bool fullTile = (rowBase + 16 <= N);    // wave-uniform
    for (int nt = 0; nt < 4; ++nt) {
        int ct = cq * 4 + nt;
        v8f c = {};
        #pragma unroll
        for (int ks = 0; ks < 4; ++ks) {
            const uint4* bp = (const uint4*)(Wp + ((size_t)(ct * 4 + ks) * 32 + lane) * 16);
            BF16Frag bf; bf.q[0] = bp[0]; bf.q[1] = bp[1];
            c = __builtin_amdgcn_wmma_f32_16x16x32_bf16(false, a[ks].v, false, bf.v,
                                                        (short)0, c, false, false);
        }
        int col = cq * 64 + nt * 16 + (lane & 15);
        float* grow = gates + (size_t)(rowBase + 8 * kh) * 256 + col;
        if (fullTile) {
            #pragma unroll
            for (int r = 0; r < 8; ++r) grow[r * 256] = c[r];
        } else {
            #pragma unroll
            for (int r = 0; r < 8; ++r)
                if (rowBase + 8 * kh + r < N) grow[r * 256] = c[r];
        }
    }
}

// ---------------- LSTM pointwise (i,f,g,o), state update, ht accumulation -----------
__global__ void k_lstm_pointwise(const float* __restrict__ gates, const float* __restrict__ bih,
                                 const float* __restrict__ bhh, float* __restrict__ cstate,
                                 float* __restrict__ hstate, float* __restrict__ ht_acc, int N) {
    int idx = blockIdx.x * blockDim.x + threadIdx.x;
    if (idx >= N * 64) return;
    int n = idx >> 6, j = idx & 63;
    const float* g = gates + (size_t)n * 256;
    float gi = g[j]       + bih[j]       + bhh[j];
    float gf = g[64 + j]  + bih[64 + j]  + bhh[64 + j];
    float gg = g[128 + j] + bih[128 + j] + bhh[128 + j];
    float go = g[192 + j] + bih[192 + j] + bhh[192 + j];
    float c = sigmoidf_(gf) * cstate[idx] + sigmoidf_(gi) * tanhf(gg);
    cstate[idx] = c;
    float hn = sigmoidf_(go) * tanhf(c);
    hstate[idx] = hn;
    ht_acc[idx] += hn;
}

// =====================================================================================
extern "C" void kernel_launch(void* const* d_in, const int* in_sizes, int n_in,
                              void* d_out, int out_size, void* d_ws, size_t ws_size,
                              hipStream_t stream) {
    const float* h        = (const float*)d_in[0];
    const int*   src      = (const int*)d_in[1];
    const int*   dst      = (const int*)d_in[2];
    const float* bn_gamma = (const float*)d_in[3];
    const float* bn_beta  = (const float*)d_in[4];
    const float* W0       = (const float*)d_in[5];
    const float* b0       = (const float*)d_in[6];
    const float* W1       = (const float*)d_in[7];
    const float* b1       = (const float*)d_in[8];
    const float* Wih      = (const float*)d_in[9];
    const float* Whh      = (const float*)d_in[10];
    const float* bih      = (const float*)d_in[11];
    const float* bhh      = (const float*)d_in[12];

    const int T = 24;
    const int N = in_sizes[0] / (T * 64);
    const int E = in_sizes[1];

    // workspace carve-out (256B aligned)
    char* base = (char*)d_ws;
    size_t cur = 0;
    auto alloc = [&](size_t bytes) -> void* {
        void* p = base + cur;
        cur = (cur + bytes + 255) & ~(size_t)255;
        return p;
    };
    float* sums   = (float*)alloc(128 * 4);
    float* scale  = (float*)alloc(64 * 4);
    float* shift  = (float*)alloc(64 * 4);
    float* onorm  = (float*)alloc((size_t)N * 4);
    float* inorm  = (float*)alloc((size_t)N * 4);
    int*   outdeg = (int*)alloc((size_t)N * 4);
    int*   indeg  = (int*)alloc((size_t)N * 4);
    int*   csroff = (int*)alloc((size_t)(N + 1) * 4);
    int*   csrcur = (int*)alloc((size_t)N * 4);
    int*   csrsrc = (int*)alloc((size_t)E * 4);
    __bf16* W0p = (__bf16*)alloc(4096 * 2);
    __bf16* W1p = (__bf16*)alloc(4096 * 2);
    __bf16* Wlp = (__bf16*)alloc(32768 * 2);
    float* mbuf   = (float*)alloc((size_t)N * 64 * 4);
    float* ybuf   = (float*)alloc((size_t)N * 64 * 4);
    float* hstate = (float*)alloc((size_t)N * 64 * 4);
    float* cstate = (float*)alloc((size_t)N * 64 * 4);
    float* gates  = (float*)alloc((size_t)N * 256 * 4);
    (void)ws_size; (void)n_in;

    // ----- zero accumulators / states / output (replay-deterministic) -----
    k_fill_f32<<<1, 256, 0, stream>>>(sums, 128, 0.f);
    k_fill_f32<<<(N + 255) / 256, 256, 0, stream>>>((float*)outdeg, N, 0.f);
    k_fill_f32<<<(N + 255) / 256, 256, 0, stream>>>((float*)indeg, N, 0.f);
    k_fill_f32<<<(N + 255) / 256, 256, 0, stream>>>((float*)csrcur, N, 0.f);
    k_fill_f32<<<2048, 256, 0, stream>>>(hstate, (long long)N * 64, 0.f);
    k_fill_f32<<<2048, 256, 0, stream>>>(cstate, (long long)N * 64, 0.f);
    k_fill_f32<<<2048, 256, 0, stream>>>((float*)d_out, out_size, 0.f);

    // ----- BatchNorm statistics -----
    k_bn_stats<<<1024, 256, 0, stream>>>(h, sums, (long long)N * T);
    k_bn_finalize<<<1, 64, 0, stream>>>(sums, bn_gamma, bn_beta, scale, shift,
                                        1.0f / (float)((long long)N * T));

    // ----- degrees, norms, CSR (dst-grouped) -----
    int eBlocks = imin((E + 255) / 256, 8192);
    k_degrees<<<eBlocks, 256, 0, stream>>>(src, dst, E, outdeg, indeg);
    k_norms<<<(N + 255) / 256, 256, 0, stream>>>(outdeg, indeg, onorm, inorm, N);
    k_scan<<<1, 256, 0, stream>>>(indeg, csroff, N);
    k_csr_fill<<<eBlocks, 256, 0, stream>>>(src, dst, E, csroff, csrcur, csrsrc);

    // ----- pack weights into WMMA-fragment layout (bf16) -----
    k_pack_w64<<<16, 256, 0, stream>>>(W0, W0p);
    k_pack_w64<<<16, 256, 0, stream>>>(W1, W1p);
    k_pack_lstm<<<128, 256, 0, stream>>>(Wih, Whh, Wlp);

    int gemmBlocks = (((N + 15) / 16) + 7) / 8;   // 8 waves/block, 16 rows/wave
    int aggBlocks  = (N + 7) / 8;                 // wave per node, 8 nodes/block
    int lstmBlocks = (N + 31) / 32;               // 32 rows/block
    int pwBlocks   = (N * 64 + 255) / 256;

    float* hs_out = (float*)d_out;                       // [T,64]
    float* ht_out = (float*)d_out + (size_t)T * 64;      // [N,64]

    for (int t = 0; t < T; ++t) {
        const float* hslice = h + (size_t)t * 64;        // row stride T*64
        // spatial layer 1: m = (BN(h_t) * onorm) @ W0
        k_gemm64<<<gemmBlocks, 256, 0, stream>>>(hslice, scale, shift, onorm, W0p, mbuf,
                                                 N, T * 64, 0);
        k_agg_gelu<<<aggBlocks, 256, 0, stream>>>(mbuf, csroff, csrsrc, inorm, b0, ybuf, N);
        // spatial layer 2: m = (y * onorm) @ W1 ; fused gather+gelu+node-mean readout
        k_gemm64<<<gemmBlocks, 256, 0, stream>>>(ybuf, scale, shift, onorm, W1p, mbuf,
                                                 N, 64, 1);
        k_agg_gelu_readout<<<aggBlocks, 256, 0, stream>>>(mbuf, csroff, csrsrc, inorm, b1,
                                                          hs_out + t * 64, N);
        // temporal: gates GEMM + pointwise LSTM update
        k_lstm_gemm<<<lstmBlocks, 256, 0, stream>>>(h, t, T, scale, shift, hstate, Wlp,
                                                    gates, N);
        k_lstm_pointwise<<<pwBlocks, 256, 0, stream>>>(gates, bih, bhh, cstate, hstate,
                                                       ht_out, N);
    }

    // finalize means
    k_scale_f32<<<6, 256, 0, stream>>>(hs_out, (long long)T * 64, 1.0f / (float)N);
    k_scale_f32<<<2048, 256, 0, stream>>>(ht_out, (long long)N * 64, 1.0f / (float)T);
}
